// GatedABCAttention_30743375905052
// MI455X (gfx1250) — compile-verified, matrix operand order checked by reading, not run on hardware
//
#include <hip/hip_runtime.h>

// ---------------------------------------------------------------------------
// Gated ABC attention, MI455X (gfx1250, wave32, WMMA bf16).
// B=2, T=2048, D=1024, H=8, M=64, KD=128.
//   1) gemm_bt (WMMA): Q/K/V = swish(X W^T), F -> (G, S) gates
//   2) scan: per (b,h) recurrence, state in registers
//   3) swish + RMSNorm
//   4) gemm_bt: out = Xn Wo^T
// GEMM: 256x64 block tile, 8 waves x (32x64), double-buffered LDS,
// float4 global loads -> packed bf16 -> 16B LDS stores -> ds_load_b128 frags.
// ---------------------------------------------------------------------------

typedef __attribute__((ext_vector_type(16))) __bf16 v16bf;
typedef __attribute__((ext_vector_type(8)))  __bf16 v8bf;
typedef __attribute__((ext_vector_type(8)))  float  v8f;

#define TILE_M 256
#define TILE_N 64
#define TILE_K 32
#define KPAD   8          // pad rows to 40 bf16 = 80 B (16B multiple, conflict-free)
#define LROW   (TILE_K + KPAD)

static __device__ __forceinline__ v8bf pack8(float4 a, float4 b) {
    v8bf p;
    p[0] = (__bf16)a.x; p[1] = (__bf16)a.y; p[2] = (__bf16)a.z; p[3] = (__bf16)a.w;
    p[4] = (__bf16)b.x; p[5] = (__bf16)b.y; p[6] = (__bf16)b.z; p[7] = (__bf16)b.w;
    return p;
}

static __device__ __forceinline__ v16bf load_frag(const __bf16* rowp, int half) {
    // lane's two contiguous 16-byte chunks of the 32-element K row
    v8bf lo = *(const v8bf*)(rowp + (half << 3));
    v8bf hi = *(const v8bf*)(rowp + 16 + (half << 3));
    return __builtin_shufflevector(lo, hi, 0, 1, 2, 3, 4, 5, 6, 7,
                                   8, 9, 10, 11, 12, 13, 14, 15);
}

// Y = act( X @ W^T ) ; X: [Nrows,Kdim] f32, W: [Nout,Kdim] f32, Y: [Nrows,Nout]
// act_mode: 0 = identity, 1 = swish, 2 = gate (Y=g=logsigmoid/16, Y2=s=1-exp(g))
__global__ __launch_bounds__(256) void gemm_bt_kernel(
    const float* __restrict__ X, const float* __restrict__ W,
    float* __restrict__ Y, float* __restrict__ Y2,
    int Nrows, int Kdim, int Nout, int act_mode)
{
    __shared__ __attribute__((aligned(16))) __bf16 As[2][TILE_M][LROW];
    __shared__ __attribute__((aligned(16))) __bf16 Bs[2][TILE_N][LROW];

    const int tid  = threadIdx.x;
    const int wave = tid >> 5;        // 0..7
    const int lane = tid & 31;
    const int half = lane >> 4;       // K-half select (0/1)
    const int l15  = lane & 15;
    const int m0   = blockIdx.y * TILE_M;
    const int n0   = blockIdx.x * TILE_N;

    v8f acc[2][4];
#pragma unroll
    for (int am = 0; am < 2; ++am)
#pragma unroll
        for (int nt = 0; nt < 4; ++nt)
#pragma unroll
            for (int j = 0; j < 8; ++j) acc[am][nt][j] = 0.0f;

    // ---- stage one K-tile (f32 -> bf16, vectorized) ----
    auto stage = [&](int buf, int k0) {
        // A: 256 rows x 32 cols -> 512 (row, 16-col-half) units, 2 per thread
#pragma unroll
        for (int it = 0; it < 2; ++it) {
            const int i = tid + (it << 8);
            const int r = i >> 1, ch = (i & 1) << 4;
            const float4* xr = (const float4*)(X + (size_t)(m0 + r) * Kdim + k0 + ch);
            float4 a0 = xr[0], a1 = xr[1], a2 = xr[2], a3 = xr[3];
            *(v8bf*)&As[buf][r][ch]     = pack8(a0, a1);
            *(v8bf*)&As[buf][r][ch + 8] = pack8(a2, a3);
        }
        // B: 64 rows x 32 cols -> 128 units, threads < 128
        if (tid < 128) {
            const int r = tid >> 1, ch = (tid & 1) << 4;
            const float4* wr = (const float4*)(W + (size_t)(n0 + r) * Kdim + k0 + ch);
            float4 b0 = wr[0], b1 = wr[1], b2 = wr[2], b3 = wr[3];
            *(v8bf*)&Bs[buf][r][ch]     = pack8(b0, b1);
            *(v8bf*)&Bs[buf][r][ch + 8] = pack8(b2, b3);
        }
    };

    stage(0, 0);
    __syncthreads();

    const int nK = Kdim / TILE_K;
    for (int kt = 0; kt < nK; ++kt) {
        const int cur = kt & 1;
        if (kt + 1 < nK) stage(cur ^ 1, (kt + 1) * TILE_K);
        // prefetch two tiles ahead into L2 (global_prefetch_b8)
        if (kt + 2 < nK)
            __builtin_prefetch(X + (size_t)(m0 + tid) * Kdim + (kt + 2) * TILE_K, 0, 1);

        // fragments: wave's 32 rows x block's 64 cols
        const int wrow = wave << 5;
        v16bf aF[2], bF[4];
#pragma unroll
        for (int am = 0; am < 2; ++am)
            aF[am] = load_frag(&As[cur][wrow + (am << 4) + l15][0], half);
#pragma unroll
        for (int nt = 0; nt < 4; ++nt)
            bF[nt] = load_frag(&Bs[cur][(nt << 4) + l15][0], half);

#pragma unroll
        for (int am = 0; am < 2; ++am)
#pragma unroll
            for (int nt = 0; nt < 4; ++nt)
                acc[am][nt] = __builtin_amdgcn_wmma_f32_16x16x32_bf16(
                    false, aF[am], false, bF[nt], (short)0, acc[am][nt], false, false);
        __syncthreads();
    }

    // store; C/D layout: VGPR r, lane<16 -> (M=r, N=lane); lane>=16 -> (M=8+r)
#pragma unroll
    for (int am = 0; am < 2; ++am) {
#pragma unroll
        for (int nt = 0; nt < 4; ++nt) {
#pragma unroll
            for (int r = 0; r < 8; ++r) {
                const int grow = m0 + (wave << 5) + (am << 4) + r + (half << 3);
                const int gcol = n0 + (nt << 4) + l15;
                const size_t idx = (size_t)grow * Nout + gcol;
                float v = acc[am][nt][r];
                if (act_mode == 0) {
                    Y[idx] = v;
                } else if (act_mode == 1) {
                    Y[idx] = v / (1.0f + __expf(-v));                       // swish
                } else {
                    float ls = fminf(v, 0.0f) - log1pf(__expf(-fabsf(v)));  // logsigmoid
                    float g  = ls * 0.0625f;                                // /GATE_NORM
                    Y[idx]  = g;
                    Y2[idx] = 1.0f - __expf(g);
                }
            }
        }
    }
}

// ---------------------------------------------------------------------------
// Sequential gated-ABC scan. One workgroup per (b,h). State in registers:
//   thread owns hk[k, m] for m = tid&63,  k in [(tid>>6)*32, +32)
//   thread owns hv[m, v] for v = tid&127, m in [(tid>>7)*32, +32)
// ---------------------------------------------------------------------------
__global__ __launch_bounds__(256) void scan_kernel(
    const float* __restrict__ Qp, const float* __restrict__ Kp,
    const float* __restrict__ Vp, const float* __restrict__ Gp,
    const float* __restrict__ Sp, float* __restrict__ Op)
{
    const int bh  = blockIdx.x;     // 0..15
    const int b   = bh >> 3;        // H = 8
    const int h   = bh & 7;
    const int tid = threadIdx.x;

    float hk[32], hv[32];
#pragma unroll
    for (int i = 0; i < 32; ++i) { hk[i] = 0.0f; hv[i] = 0.0f; }

    const int m_hk  = tid & 63;
    const int kbase = (tid >> 6) << 5;   // 0,32,64,96
    const int v_hv  = tid & 127;
    const int mbase = (tid >> 7) << 5;   // 0,32

    __shared__ float qv[128], kv[128], vv[128];
    __shared__ float ge[64], sv[64], pz[64];
    __shared__ float part[4][64];
    __shared__ float part2[2][128];

    const float scale = 0.08838834764831845f; // 128^-0.5
    const int D = 1024, F = 512;

    for (int t = 0; t < 2048; ++t) {
        const size_t n = (size_t)(b * 2048 + t);
        // ---- load step inputs ----
        if (tid < 128) {
            qv[tid] = Qp[n * D + h * 128 + tid] * scale;
            kv[tid] = Kp[n * D + h * 128 + tid];
            vv[tid] = Vp[n * D + h * 128 + tid];
        } else if (tid < 192) {
            int m = tid - 128;
            ge[m] = __expf(Gp[n * F + h * 64 + m]);
        } else {
            int m = tid - 192;
            sv[m] = Sp[n * F + h * 64 + m];
        }
        __syncthreads();

        // ---- hk update + partial q.hk ----
        {
            const float gem = ge[m_hk];
            const float sm  = sv[m_hk];
            float acc = 0.0f;
#pragma unroll
            for (int i = 0; i < 32; ++i) {
                const int k = kbase + i;
                const float nh = hk[i] * gem + kv[k] * sm;
                hk[i] = nh;
                acc += qv[k] * nh;
            }
            part[tid >> 6][m_hk] = acc;
        }
        // ---- hv update (independent of softmax) ----
        {
            const float vval = vv[v_hv];
#pragma unroll
            for (int i = 0; i < 32; ++i) {
                const int m = mbase + i;
                hv[i] = hv[i] * ge[m] + sv[m] * vval;
            }
        }
        __syncthreads();

        // ---- softmax over 64 slots, wave 0 (2 slots / lane) ----
        if (tid < 32) {
            float z0 = part[0][tid]      + part[1][tid]      + part[2][tid]      + part[3][tid];
            float z1 = part[0][tid + 32] + part[1][tid + 32] + part[2][tid + 32] + part[3][tid + 32];
            float mx = fmaxf(z0, z1);
#pragma unroll
            for (int off = 16; off > 0; off >>= 1) mx = fmaxf(mx, __shfl_xor(mx, off, 32));
            float e0 = __expf(z0 - mx), e1 = __expf(z1 - mx);
            float sm = e0 + e1;
#pragma unroll
            for (int off = 16; off > 0; off >>= 1) sm += __shfl_xor(sm, off, 32);
            float inv = 1.0f / sm;
            pz[tid]      = e0 * inv;
            pz[tid + 32] = e1 * inv;
        }
        __syncthreads();

        // ---- o = p . hv ----
        {
            float acc = 0.0f;
#pragma unroll
            for (int i = 0; i < 32; ++i) acc += pz[mbase + i] * hv[i];
            part2[tid >> 7][v_hv] = acc;
        }
        __syncthreads();
        if (tid < 128)
            Op[n * D + h * 128 + tid] = part2[0][tid] + part2[1][tid];
    }
}

// ---------------------------------------------------------------------------
// y = RMSNorm(swish(x)) * g_weight     (one block per row of 1024)
// ---------------------------------------------------------------------------
__global__ __launch_bounds__(256) void swish_rms_kernel(
    const float* __restrict__ X, const float* __restrict__ w,
    float* __restrict__ Y)
{
    const int row = blockIdx.x;
    const float* xr = X + (size_t)row * 1024;
    float loc[4];
    float ss = 0.0f;
#pragma unroll
    for (int i = 0; i < 4; ++i) {
        const float x  = xr[threadIdx.x + 256 * i];
        const float sw = x / (1.0f + __expf(-x));
        loc[i] = sw;
        ss += sw * sw;
    }
#pragma unroll
    for (int off = 16; off > 0; off >>= 1) ss += __shfl_xor(ss, off, 32);
    __shared__ float wred[8];
    if ((threadIdx.x & 31) == 0) wred[threadIdx.x >> 5] = ss;
    __syncthreads();
    float tot = 0.0f;
#pragma unroll
    for (int i = 0; i < 8; ++i) tot += wred[i];
    const float r = rsqrtf(tot * (1.0f / 1024.0f) + 1e-5f);
#pragma unroll
    for (int i = 0; i < 4; ++i) {
        const int c = threadIdx.x + 256 * i;
        Y[(size_t)row * 1024 + c] = loc[i] * r * w[c];
    }
}

// ---------------------------------------------------------------------------
extern "C" void kernel_launch(void* const* d_in, const int* in_sizes, int n_in,
                              void* d_out, int out_size, void* d_ws, size_t ws_size,
                              hipStream_t stream) {
    const float* X  = (const float*)d_in[0];   // [4096,1024]
    const float* Wq = (const float*)d_in[1];
    const float* Wk = (const float*)d_in[2];
    const float* Wv = (const float*)d_in[3];
    const float* Wf = (const float*)d_in[4];   // [512,1024]
    const float* Wo = (const float*)d_in[5];
    const float* gw = (const float*)d_in[6];   // [1024]

    const size_t ND = 4096ull * 1024ull;
    const size_t NF = 4096ull * 512ull;
    float* ws  = (float*)d_ws;
    float* Q   = ws;             // 16 MB
    float* Kt  = Q   + ND;       // 16 MB
    float* Vt  = Kt  + ND;       // 16 MB
    float* G   = Vt  + ND;       //  8 MB
    float* Sv  = G   + NF;       //  8 MB
    float* Oin = Sv  + NF;       // 16 MB
    float* Xn  = Q;              // reuse Q after the scan

    dim3 blk(256);
    dim3 gQKV(1024 / TILE_N, 4096 / TILE_M);   // 16 x 16
    dim3 gF(512 / TILE_N, 4096 / TILE_M);      //  8 x 16

    gemm_bt_kernel<<<gQKV, blk, 0, stream>>>(X, Wq, Q,  nullptr, 4096, 1024, 1024, 1);
    gemm_bt_kernel<<<gQKV, blk, 0, stream>>>(X, Wk, Kt, nullptr, 4096, 1024, 1024, 1);
    gemm_bt_kernel<<<gQKV, blk, 0, stream>>>(X, Wv, Vt, nullptr, 4096, 1024, 1024, 1);
    gemm_bt_kernel<<<gF,   blk, 0, stream>>>(X, Wf, G,  Sv,      4096, 1024,  512, 2);

    scan_kernel<<<16, blk, 0, stream>>>(Q, Kt, Vt, G, Sv, Oin);

    swish_rms_kernel<<<4096, blk, 0, stream>>>(Oin, gw, Xn);

    gemm_bt_kernel<<<gQKV, blk, 0, stream>>>(Xn, Wo, (float*)d_out, nullptr, 4096, 1024, 1024, 0);
}
